// Head_3616362463318
// MI455X (gfx1250) — compile-verified, matrix operand order checked by reading, not run on hardware
//
#include <hip/hip_runtime.h>
#include <hip/hip_bf16.h>

typedef __bf16 bf16;
typedef __attribute__((ext_vector_type(16))) __bf16 v16bf;
typedef __attribute__((ext_vector_type(8)))  __bf16 v8bf;   // 16 bytes
typedef __attribute__((ext_vector_type(8)))  float  v8f;
typedef __attribute__((ext_vector_type(4)))  float  v4f;

constexpr int Bz = 8;        // batch
constexpr int T  = 4096;     // sequence
constexpr int C  = 1024;     // n_emb
constexpr int H  = 128;      // head size
constexpr int M  = Bz * T;   // flattened rows
constexpr int XP = 1032;     // padded LDS row stride (bf16) for x strip
constexpr int PP = 40;       // padded LDS row stride (bf16) for P tile

// per-wave LDS RAW fence: LDS is in-order within a wave (ISA 7.3), only the
// dscnt wait for returned data is needed; "memory" stops compile-time reorder.
__device__ __forceinline__ void lds_fence() {
    asm volatile("s_wait_dscnt 0x0" ::: "memory");
}
__device__ __forceinline__ void compiler_fence() {
    asm volatile("" ::: "memory");
}

// ---- fragment loader (16-bit A/B layout, CDNA5 ISA 7.12.2) ----
// lane = (m = lane&15, half = lane>>4); elements 0..7 -> k = half*8 + i,
// elements 8..15 -> k = 16 + half*8 + i. Two contiguous 16-byte chunks.
__device__ __forceinline__ v16bf load_frag_bf16(const bf16* p, int half) {
    v8bf lo = *reinterpret_cast<const v8bf*>(p + half * 8);
    v8bf hi = *reinterpret_cast<const v8bf*>(p + 16 + half * 8);
    v16bf r;
#pragma unroll
    for (int i = 0; i < 8; ++i) { r[i] = lo[i]; r[8 + i] = hi[i]; }
    return r;
}

// =====================================================================
// Kernel 0: fp32 -> bf16 conversion (used for the 3 weight matrices)
// =====================================================================
__global__ __launch_bounds__(256) void cvt_f32_bf16_kernel(
    const float* __restrict__ src, bf16* __restrict__ dst, int n) {
    int i = (blockIdx.x * 256 + threadIdx.x) * 8;
    if (i >= n) return;
    const v4f* p = reinterpret_cast<const v4f*>(src + i);
    v4f c0 = p[0], c1 = p[1];
    v8bf pk;
#pragma unroll
    for (int j = 0; j < 4; ++j) { pk[j] = (bf16)c0[j]; pk[4 + j] = (bf16)c1[j]; }
    *reinterpret_cast<v8bf*>(dst + i) = pk;
}

// =====================================================================
// Kernel 1: fused QKV projection. 3 waves per block; the block stages its
// 16x1024 x-strip into LDS (f32->bf16 once), then wave w computes
// projection w (0=Q,1=K,2=V) entirely from LDS A-fragments + bf16 W.
// Q,K stored bf16 [B*T, H]; V stored bf16 TRANSPOSED [B, H, T].
// =====================================================================
__global__ __launch_bounds__(96) void qkv_proj_kernel(
    const float* __restrict__ x, const bf16* __restrict__ Wb,
    bf16* __restrict__ Qb, bf16* __restrict__ Kb, bf16* __restrict__ Vt) {

    __shared__ bf16 xs[16 * XP];           // padded: conflict-free b128 reads

    const int tid   = threadIdx.x;
    const int strip = blockIdx.x;          // 16-row strip of flattened [B*T]

    // ---- cooperative stage: 16x1024 f32 -> bf16 into LDS ----
    const float* xsrc = x + (size_t)strip * 16 * C;
    for (int idx = tid; idx < 2048; idx += 96) {   // 2048 chunks of 8 elems
        int row = idx >> 7;                        // idx / 128
        int c8  = (idx & 127) * 8;
        const v4f* p = reinterpret_cast<const v4f*>(xsrc + (size_t)row * C + c8);
        v4f c0 = p[0], c1 = p[1];
        v8bf pk;
#pragma unroll
        for (int j = 0; j < 4; ++j) { pk[j] = (bf16)c0[j]; pk[4 + j] = (bf16)c1[j]; }
        *reinterpret_cast<v8bf*>(&xs[row * XP + c8]) = pk;
    }
    __syncthreads();

    const int wave = tid >> 5;             // 0=Q 1=K 2=V
    const int lane = tid & 31;
    const int m    = lane & 15;
    const int half = lane >> 4;
    const bf16* W  = Wb + (size_t)wave * H * C;

    v8f acc[8] = {};                       // 8 n-tiles cover H=128
#pragma unroll 2
    for (int kt = 0; kt < C; kt += 32) {
        v16bf a = load_frag_bf16(&xs[m * XP + kt], half);   // ds_load_b128 x2
#pragma unroll
        for (int nt = 0; nt < 8; ++nt) {
            // B[k=c, n=h] = W[h, c]; lane column n = nt*16 + m
            const bf16* wrow = W + (size_t)(nt * 16 + m) * C + kt;
            v16bf bfrag = load_frag_bf16(wrow, half);
            acc[nt] = __builtin_amdgcn_wmma_f32_16x16x32_bf16(
                false, a, false, bfrag, (short)0, acc[nt], false, false);
        }
    }

    if (wave < 2) {
        bf16* outb = (wave == 0) ? Qb : Kb;
#pragma unroll
        for (int nt = 0; nt < 8; ++nt)
#pragma unroll
            for (int v = 0; v < 8; ++v) {
                int r = strip * 16 + v + 8 * half;          // D row = v + 8*half
                outb[(size_t)r * H + nt * 16 + m] = (bf16)acc[nt][v];
            }
    } else {
        // Vt[b][h][t]: lane column n = h fixed, 8 consecutive t -> packed store
        const int b  = (strip * 16) / T;
        const int t0 = (strip * 16) % T + 8 * half;
#pragma unroll
        for (int nt = 0; nt < 8; ++nt) {
            int h = nt * 16 + m;
            v8bf pk;
#pragma unroll
            for (int v = 0; v < 8; ++v) pk[v] = (bf16)acc[nt][v];
            *reinterpret_cast<v8bf*>(Vt + ((size_t)b * H + h) * T + t0) = pk;
        }
    }
}

// =====================================================================
// Kernel 2: causal flash attention. 128 threads = 4 independent waves,
// each owning one 16-row Q tile (different trip counts -> no block sync;
// per-wave LDS region + s_wait_dscnt fence instead).
// =====================================================================
__global__ __launch_bounds__(128) void attn_kernel(
    const bf16* __restrict__ Qb, const bf16* __restrict__ Kb,
    const bf16* __restrict__ Vt, float* __restrict__ out) {

    __shared__ bf16 Plds[4 * 16 * PP];

    const int tid  = threadIdx.x;
    const int wave = tid >> 5;
    const int lane = tid & 31;
    const int m    = lane & 15;
    const int half = lane >> 4;
    const int qt   = blockIdx.x * 4 + wave;   // 16-row tile within batch
    const int b    = blockIdx.y;
    bf16* myP = &Plds[wave * 16 * PP];
    const float scale = 0.08838834764831845f; // 1/sqrt(128)

    // Q A-fragments: 4 fragments over k = h (0..127)
    v16bf qf[4];
    const bf16* Qrow = Qb + ((size_t)b * T + qt * 16 + m) * H;
#pragma unroll
    for (int kk = 0; kk < 4; ++kk) qf[kk] = load_frag_bf16(Qrow + kk * 32, half);

    v8f o[8] = {};
    float mstat[8], lsum[8];
#pragma unroll
    for (int v = 0; v < 8; ++v) { mstat[v] = -__builtin_inff(); lsum[v] = 0.f; }

    const int nchunk = (qt * 16 + 16 + 31) >> 5;
    for (int sc = 0; sc < nchunk; ++sc) {
        const int s_base = sc * 32;
        const bf16* K0 = Kb + ((size_t)b * T + s_base + m) * H;
        const bf16* K1 = K0 + 16 * H;

        if (sc + 1 < nchunk) {            // global_prefetch_b8 on next chunk
            __builtin_prefetch(K0 + 32 * H, 0, 1);
            __builtin_prefetch(Vt + ((size_t)b * H + m) * T + s_base + 32, 0, 1);
        }

        // ---- S = Q K^T : two 16x16 tiles, K-dim = H -> 4 wmma each ----
        v8f s0 = {}, s1 = {};
#pragma unroll
        for (int kk = 0; kk < 4; ++kk) {
            v16bf b0 = load_frag_bf16(K0 + kk * 32, half);
            s0 = __builtin_amdgcn_wmma_f32_16x16x32_bf16(
                false, qf[kk], false, b0, (short)0, s0, false, false);
            v16bf b1 = load_frag_bf16(K1 + kk * 32, half);
            s1 = __builtin_amdgcn_wmma_f32_16x16x32_bf16(
                false, qf[kk], false, b1, (short)0, s1, false, false);
        }

        // ---- causal mask + online softmax; row r = v + 8*half, col = m ----
        const int c0 = s_base + m, c1 = s_base + 16 + m;
#pragma unroll
        for (int v = 0; v < 8; ++v) {
            const int tg = qt * 16 + v + 8 * half;
            float e0 = (c0 <= tg) ? s0[v] * scale : -__builtin_inff();
            float e1 = (c1 <= tg) ? s1[v] * scale : -__builtin_inff();

            float mx = fmaxf(e0, e1);
#pragma unroll
            for (int off = 1; off < 16; off <<= 1)
                mx = fmaxf(mx, __shfl_xor(mx, off, 32));   // 16-lane group

            float mnew  = fmaxf(mstat[v], mx);
            float alpha = __expf(mstat[v] - mnew);
            float p0 = __expf(e0 - mnew);
            float p1 = __expf(e1 - mnew);
            float rs = p0 + p1;
#pragma unroll
            for (int off = 1; off < 16; off <<= 1)
                rs += __shfl_xor(rs, off, 32);

            lsum[v]  = lsum[v] * alpha + rs;
            mstat[v] = mnew;
#pragma unroll
            for (int nt = 0; nt < 8; ++nt) o[nt][v] *= alpha;

            // stage P (C-layout -> LDS [16][PP]) as bf16
            myP[(v + 8 * half) * PP + m]      = (bf16)p0;
            myP[(v + 8 * half) * PP + 16 + m] = (bf16)p1;
        }
        lds_fence();                       // RAW: stores visible before reads

        // ---- read P back in A-fragment layout (conflict-free, PP=40) ----
        v16bf pf;
        {
            v8bf lo = *reinterpret_cast<const v8bf*>(&myP[m * PP + half * 8]);
            v8bf hi = *reinterpret_cast<const v8bf*>(&myP[m * PP + 16 + half * 8]);
#pragma unroll
            for (int i = 0; i < 8; ++i) { pf[i] = lo[i]; pf[8 + i] = hi[i]; }
        }
        compiler_fence();                  // keep next iter's stores below reads

        // ---- O += P V : B[k=s, n=h] = Vt[h][s_base + k] ----
#pragma unroll
        for (int nt = 0; nt < 8; ++nt) {
            const bf16* Vrow = Vt + ((size_t)b * H + nt * 16 + m) * T + s_base;
            v16bf vf = load_frag_bf16(Vrow, half);
            o[nt] = __builtin_amdgcn_wmma_f32_16x16x32_bf16(
                false, pf, false, vf, (short)0, o[nt], false, false);
        }
    }

    // ---- normalize and write fp32 output [B, T, H] ----
#pragma unroll
    for (int nt = 0; nt < 8; ++nt)
#pragma unroll
        for (int v = 0; v < 8; ++v) {
            int t = qt * 16 + v + 8 * half;
            out[((size_t)b * T + t) * H + nt * 16 + m] = o[nt][v] / lsum[v];
        }
}

extern "C" void kernel_launch(void* const* d_in, const int* in_sizes, int n_in,
                              void* d_out, int out_size, void* d_ws, size_t ws_size,
                              hipStream_t stream) {
    const float* x  = (const float*)d_in[0];
    const float* Wq = (const float*)d_in[1];
    const float* Wk = (const float*)d_in[2];
    const float* Wv = (const float*)d_in[3];
    float* out = (float*)d_out;

    // workspace: Wb bf16 [3,H,C], Q [B*T,H], K [B*T,H], V^T [B,H,T] (~26 MB)
    bf16* Wb = (bf16*)d_ws;
    bf16* Qb = Wb + (size_t)3 * H * C;
    bf16* Kb = Qb + (size_t)M * H;
    bf16* Vt = Kb + (size_t)M * H;

    const int nW = H * C;                           // 131072
    dim3 gridW((nW / 8 + 255) / 256);
    cvt_f32_bf16_kernel<<<gridW, 256, 0, stream>>>(Wq, Wb,             nW);
    cvt_f32_bf16_kernel<<<gridW, 256, 0, stream>>>(Wk, Wb + nW,        nW);
    cvt_f32_bf16_kernel<<<gridW, 256, 0, stream>>>(Wv, Wb + 2 * nW,    nW);

    qkv_proj_kernel<<<dim3(M / 16), 96, 0, stream>>>(x, Wb, Qb, Kb, Vt);

    attn_kernel<<<dim3(T / 64, Bz), 128, 0, stream>>>(Qb, Kb, Vt, out);
}
